// PointNetSaModule_52604759441832
// MI455X (gfx1250) — matrix operand order repaired and checked by reading, compiled, not run on hardware
//
#include <hip/hip_runtime.h>
#include <hip/hip_bf16.h>

// ---------------------------------------------------------------------------
// PointNet SA module, fused: gather -> (67->64->64->128 MLP, BN folded) -> maxK
// bf16 WMMA (v_wmma_f32_16x16x32_bf16), wave32, 4 waves/block, each wave owns
// one sampled point's 16 neighbor rows (max over K = column max of its strip).
// Weights are BN-folded ONCE by a prep kernel into d_ws as a ready-to-use LDS
// image; the main kernel pulls it into LDS with global_load_async_to_lds_b128.
// ---------------------------------------------------------------------------

typedef __attribute__((ext_vector_type(16))) __bf16  v16bf;
typedef __attribute__((ext_vector_type(8)))  float   v8f;
typedef __attribute__((ext_vector_type(4)))  unsigned int u32x4;
typedef __attribute__((ext_vector_type(4)))  float   f32x4;

#define BATCH    4
#define HWTOT    65536      // H*W
#define NSAMP    16384      // h*w
#define KNB      16
#define CIN      64

#define A0_STRIDE  104      // 96 (padded K for layer0) + 8 pad, 16B-aligned rows
#define ACT_STRIDE 72       // 64 + 8 pad, 144B rows (16B aligned)

// Folded-weight image layout (ushort elements unless noted):
//   [0,6144)        sW0t  (64 rows n x 96 k)
//   [6144,10240)    sW1t  (64 x 64)
//   [10240,18432)   sW2t  (128 x 64)
//   byte 36864..37888  biases: f32 bias0[64] bias1[64] bias2[128]
//   byte 37888..38912  zero pad (copy granularity)
#define WIMG_USHORTS 18432
#define WIMG_BYTES   38912          // 19 * 128 threads * 16B
#define WCOPY_CHUNKS 19

__device__ __forceinline__ unsigned short f2bf(float x) {
  union { float f; unsigned u; } v; v.f = x;
  unsigned r = v.u + 0x7fffu + ((v.u >> 16) & 1u);   // round-to-nearest-even
  return (unsigned short)(r >> 16);
}

union FragU { v16bf v; u32x4 q[2]; };

// A fragment (16x32 bf16): lane holds row m=lane&15; sub=lane>>4 selects the
// K sub-range. Caller passes k0 = Kchunk + sub*8. q[0]=K0..K0+7, q[1]=+16.
__device__ __forceinline__ v16bf load_a(const unsigned short* rowBase, int k0) {
  FragU f;
  f.q[0] = *(const u32x4*)(rowBase + k0);
  f.q[1] = *(const u32x4*)(rowBase + k0 + 16);
  return f.v;
}

// B fragment (32x16 bf16): lane holds column n=lane&15; sub=lane>>4 selects
// K sub-range (k0 = Kchunk + sub*8; q[0]=K0..K0+7, q[1]=+16). Weights stored n-major (transposed) so K is contiguous.
__device__ __forceinline__ v16bf load_b(const unsigned short* colBase, int k0) {
  FragU f;
  f.q[0] = *(const u32x4*)(colBase + k0);
  f.q[1] = *(const u32x4*)(colBase + k0 + 16);
  return f.v;
}

__device__ __forceinline__ v8f wmma_bf16(v16bf a, v16bf b, v8f c) {
  return __builtin_amdgcn_wmma_f32_16x16x32_bf16(false, a, false, b,
                                                 (short)0, c, false, false);
}

// Async 16B global -> LDS copy (tracked by ASYNCcnt), GV addressing mode.
__device__ __forceinline__ void async_copy_b128(unsigned ldsOff,
                                                unsigned long long gaddr) {
  asm volatile("global_load_async_to_lds_b128 %0, %1, off"
               :: "v"(ldsOff), "v"(gaddr) : "memory");
}
__device__ __forceinline__ void wait_async0() {
  asm volatile("s_wait_asynccnt 0x0" ::: "memory");
}

// ------------------------- prep: fold BN once ------------------------------
__global__ __launch_bounds__(256)
void fold_weights_kernel(
    const float* __restrict__ w0, const float* __restrict__ b0,
    const float* __restrict__ g0, const float* __restrict__ be0,
    const float* __restrict__ m0, const float* __restrict__ v0,
    const float* __restrict__ w1, const float* __restrict__ b1,
    const float* __restrict__ g1, const float* __restrict__ be1,
    const float* __restrict__ m1, const float* __restrict__ v1,
    const float* __restrict__ w2, const float* __restrict__ b2,
    const float* __restrict__ g2, const float* __restrict__ be2,
    const float* __restrict__ m2, const float* __restrict__ v2,
    unsigned short* __restrict__ wimg)
{
  unsigned short* W0t = wimg;
  unsigned short* W1t = wimg + 64 * 96;
  unsigned short* W2t = wimg + 64 * 96 + 64 * 64;
  float* bias = (float*)(wimg + WIMG_USHORTS);    // 256 f32
  const int t = threadIdx.x;
  if (t < 64) {
    const int n = t;
    {
      float s = g0[n] * rsqrtf(v0[n] + 1e-5f);
      bias[n] = (b0[n] - m0[n]) * s + be0[n];
      // channel reorder: K 0..63 = point channels (orig rows 3..66),
      // K 64..66 = xyz (orig rows 0..2), K 67..95 = zero pad.
      for (int k = 0; k < 64; ++k) W0t[n * 96 + k]      = f2bf(w0[(3 + k) * 64 + n] * s);
      for (int k = 0; k < 3;  ++k) W0t[n * 96 + 64 + k] = f2bf(w0[k * 64 + n] * s);
      for (int k = 67; k < 96; ++k) W0t[n * 96 + k] = 0;
    }
    {
      float s = g1[n] * rsqrtf(v1[n] + 1e-5f);
      bias[64 + n] = (b1[n] - m1[n]) * s + be1[n];
      for (int k = 0; k < 64; ++k) W1t[n * 64 + k] = f2bf(w1[k * 64 + n] * s);
    }
  } else if (t < 192) {
    const int nn = t - 64;                     // 0..127
    float s = g2[nn] * rsqrtf(v2[nn] + 1e-5f);
    bias[128 + nn] = (b2[nn] - m2[nn]) * s + be2[nn];
    for (int k = 0; k < 64; ++k) W2t[nn * 64 + k] = f2bf(w2[k * 128 + nn] * s);
  } else {
    // zero the copy-granularity tail [37888, 38912)
    unsigned* p = (unsigned*)((char*)wimg + WIMG_BYTES - 1024);
    const int i0 = (t - 192) * 4;
    for (int i = 0; i < 4; ++i) p[i0 + i] = 0u;
  }
}

// ------------------------------ main kernel --------------------------------
template <bool SELF_FOLD>
__global__ __launch_bounds__(128)
void pointnet_sa_kernel(
    const float* __restrict__ xyz_proj,        // (B,H,W,3)
    const float* __restrict__ points_proj,     // (B,H,W,64)
    const float* __restrict__ xyz_sampled,     // (B,h,w,3)
    const int*   __restrict__ neighbor_idx,    // (B, n*K)
    const float* __restrict__ valid_mask,      // (B,n,K,1)
    const float* __restrict__ w0, const float* __restrict__ b0,
    const float* __restrict__ g0, const float* __restrict__ be0,
    const float* __restrict__ m0, const float* __restrict__ v0,
    const float* __restrict__ w1, const float* __restrict__ b1,
    const float* __restrict__ g1, const float* __restrict__ be1,
    const float* __restrict__ m1, const float* __restrict__ v1,
    const float* __restrict__ w2, const float* __restrict__ b2,
    const float* __restrict__ g2, const float* __restrict__ be2,
    const float* __restrict__ m2, const float* __restrict__ v2,
    const unsigned short* __restrict__ foldedW,   // d_ws image (or null)
    float* __restrict__ out)
{
  __shared__ __align__(16) unsigned char sWimg[WIMG_BYTES];
  __shared__ __align__(16) unsigned short sA[64 * A0_STRIDE];     // 13.0 KB
  __shared__ __align__(16) unsigned short sAct1[64 * ACT_STRIDE]; //  9.0 KB

  unsigned short* sW0 = (unsigned short*)sWimg;
  unsigned short* sW1 = sW0 + 64 * 96;
  unsigned short* sW2 = sW1 + 64 * 64;
  float* sBias0 = (float*)(sW0 + WIMG_USHORTS);
  float* sBias1 = sBias0 + 64;
  float* sBias2 = sBias1 + 64;

  const int t = threadIdx.x;

  if (!SELF_FOLD) {
    // ---- pull the folded-weight image into LDS with async b128 copies ----
    const unsigned ldsBase = (unsigned)(unsigned long long)(void*)sWimg;
    const unsigned long long src = (unsigned long long)foldedW;
    #pragma unroll
    for (int c = 0; c < WCOPY_CHUNKS; ++c) {
      const unsigned off = (unsigned)((c * 128 + t) * 16);
      async_copy_b128(ldsBase + off, src + off);
    }
  } else {
    // ---- fallback: fold BN locally (wave-uniform branches) ----
    if (t < 64) {
      const int n = t;
      {
        float s = g0[n] * rsqrtf(v0[n] + 1e-5f);
        sBias0[n] = (b0[n] - m0[n]) * s + be0[n];
        #pragma unroll 4
        for (int k = 0; k < 64; ++k) sW0[n * 96 + k]      = f2bf(w0[(3 + k) * 64 + n] * s);
        for (int k = 0; k < 3;  ++k) sW0[n * 96 + 64 + k] = f2bf(w0[k * 64 + n] * s);
        for (int k = 67; k < 96; ++k) sW0[n * 96 + k] = 0;
      }
      {
        float s = g1[n] * rsqrtf(v1[n] + 1e-5f);
        sBias1[n] = (b1[n] - m1[n]) * s + be1[n];
        #pragma unroll 4
        for (int k = 0; k < 64; ++k) sW1[n * 64 + k] = f2bf(w1[k * 64 + n] * s);
      }
    } else {
      #pragma unroll
      for (int rep = 0; rep < 2; ++rep) {
        const int nn = (t - 64) + rep * 64;
        float s = g2[nn] * rsqrtf(v2[nn] + 1e-5f);
        sBias2[nn] = (b2[nn] - m2[nn]) * s + be2[nn];
        #pragma unroll 4
        for (int k = 0; k < 64; ++k) sW2[nn * 64 + k] = f2bf(w2[k * 128 + nn] * s);
      }
    }
  }

  // -------- gather: 64 rows (4 points x 16 neighbors), 2 threads/row --------
  {
    const int row  = t >> 1;           // 0..63
    const int half = t & 1;            // channel half
    const int pl   = row >> 4;         // point-in-block
    const int kk   = row & 15;         // neighbor
    const long gp  = (long)blockIdx.x * 4 + pl;         // global sampled point
    const int  b   = (int)(gp >> 14);                   // /16384
    const int  nl  = (int)(gp & 16383);
    const long ni  = (long)b * (NSAMP * KNB) + (long)nl * KNB + kk;
    const int  idx = neighbor_idx[ni];
    const float vm = valid_mask[ni];

    unsigned short* arow = &sA[row * A0_STRIDE];
    const float* psrc = points_proj + ((long)b * HWTOT + idx) * CIN + half * 32;
    #pragma unroll
    for (int c = 0; c < 32; c += 4) {
      f32x4 p = *(const f32x4*)(psrc + c);
      const int o = half * 32 + c;
      arow[o + 0] = f2bf(p.x * vm);
      arow[o + 1] = f2bf(p.y * vm);
      arow[o + 2] = f2bf(p.z * vm);
      arow[o + 3] = f2bf(p.w * vm);
    }
    if (half) {
      const float* xg = xyz_proj    + ((long)b * HWTOT + idx) * 3;
      const float* xs = xyz_sampled + ((long)b * NSAMP + nl) * 3;
      arow[64] = f2bf(xg[0] * vm - xs[0]);
      arow[65] = f2bf(xg[1] * vm - xs[1]);
      arow[66] = f2bf(xg[2] * vm - xs[2]);
      #pragma unroll
      for (int k = 67; k < 96; ++k) arow[k] = 0;
    }
  }
  if (!SELF_FOLD) wait_async0();   // own async LDS writes done before barrier
  __syncthreads();

  // -------- WMMA MLP: each wave owns rows [16*wave, 16*wave+16) -------------
  const int wave = t >> 5;
  const int lane = t & 31;
  const int ln   = lane & 15;
  const int sub  = lane >> 4;
  const v8f vzero = {0.f, 0.f, 0.f, 0.f, 0.f, 0.f, 0.f, 0.f};

  // ---- layer 0: (16x96) x (96x64) ----
  v8f acc0[4] = {vzero, vzero, vzero, vzero};
  const unsigned short* aB0 = &sA[(wave * 16 + ln) * A0_STRIDE];
  #pragma unroll
  for (int kc = 0; kc < 3; ++kc) {
    v16bf a = load_a(aB0, kc * 32 + sub * 8);
    #pragma unroll
    for (int ct = 0; ct < 4; ++ct) {
      v16bf bb = load_b(&sW0[(ct * 16 + ln) * 96], kc * 32 + sub * 8);
      acc0[ct] = wmma_bf16(a, bb, acc0[ct]);
    }
  }
  #pragma unroll
  for (int ct = 0; ct < 4; ++ct) {
    const int col = ct * 16 + ln;
    const float bias = sBias0[col];
    #pragma unroll
    for (int i = 0; i < 8; ++i) {
      float y = fmaxf(acc0[ct][i] + bias, 0.f);
      sAct1[(wave * 16 + sub * 8 + i) * ACT_STRIDE + col] = f2bf(y);
    }
  }

  // ---- layer 1: (16x64) x (64x64) ----
  v8f acc1[4] = {vzero, vzero, vzero, vzero};
  const unsigned short* aB1 = &sAct1[(wave * 16 + ln) * ACT_STRIDE];
  #pragma unroll
  for (int kc = 0; kc < 2; ++kc) {
    v16bf a = load_a(aB1, kc * 32 + sub * 8);
    #pragma unroll
    for (int ct = 0; ct < 4; ++ct) {
      v16bf bb = load_b(&sW1[(ct * 16 + ln) * 64], kc * 32 + sub * 8);
      acc1[ct] = wmma_bf16(a, bb, acc1[ct]);
    }
  }
  unsigned short* sAct2 = sA;   // reuse layer-0 buffer (ACT_STRIDE layout)
  #pragma unroll
  for (int ct = 0; ct < 4; ++ct) {
    const int col = ct * 16 + ln;
    const float bias = sBias1[col];
    #pragma unroll
    for (int i = 0; i < 8; ++i) {
      float y = fmaxf(acc1[ct][i] + bias, 0.f);
      sAct2[(wave * 16 + sub * 8 + i) * ACT_STRIDE + col] = f2bf(y);
    }
  }

  // ---- layer 2: (16x64) x (64x128) ----
  v8f acc2[8] = {vzero, vzero, vzero, vzero, vzero, vzero, vzero, vzero};
  const unsigned short* aB2 = &sAct2[(wave * 16 + ln) * ACT_STRIDE];
  #pragma unroll
  for (int kc = 0; kc < 2; ++kc) {
    v16bf a = load_a(aB2, kc * 32 + sub * 8);
    #pragma unroll
    for (int ct = 0; ct < 8; ++ct) {
      v16bf bb = load_b(&sW2[(ct * 16 + ln) * 64], kc * 32 + sub * 8);
      acc2[ct] = wmma_bf16(a, bb, acc2[ct]);
    }
  }

  // ---- bias + ReLU + max over the wave's 16 rows (= K neighbors) ----------
  const long gp = (long)blockIdx.x * 4 + wave;     // global sampled point
  float* o1 = out + gp * 128;                                  // (B,n,128)
  float* o2 = out + (long)BATCH * NSAMP * 128 + gp * 128;      // proj copy
  #pragma unroll
  for (int ct = 0; ct < 8; ++ct) {
    const int col = ct * 16 + ln;
    const float bias = sBias2[col];
    float mx = 0.f;                       // ReLU guarantees >= 0
    #pragma unroll
    for (int i = 0; i < 8; ++i) mx = fmaxf(mx, acc2[ct][i] + bias);
    mx = fmaxf(mx, __shfl_xor(mx, 16, 32));   // fold rows 0..7 with 8..15
    if (sub == 0) { o1[col] = mx; o2[col] = mx; }
  }
}

extern "C" void kernel_launch(void* const* d_in, const int* in_sizes, int n_in,
                              void* d_out, int out_size, void* d_ws, size_t ws_size,
                              hipStream_t stream) {
  (void)in_sizes; (void)n_in; (void)out_size;
  const float* xyz_proj     = (const float*)d_in[0];
  const float* points_proj  = (const float*)d_in[1];
  const float* xyz_sampled  = (const float*)d_in[2];
  const int*   neighbor_idx = (const int*)  d_in[3];
  const float* valid_mask   = (const float*)d_in[4];
  const float* w0 = (const float*)d_in[5];
  const float* b0 = (const float*)d_in[6];
  const float* g0 = (const float*)d_in[7];
  const float* be0 = (const float*)d_in[8];
  const float* m0 = (const float*)d_in[9];
  const float* v0 = (const float*)d_in[10];
  const float* w1 = (const float*)d_in[11];
  const float* b1 = (const float*)d_in[12];
  const float* g1 = (const float*)d_in[13];
  const float* be1 = (const float*)d_in[14];
  const float* m1 = (const float*)d_in[15];
  const float* v1 = (const float*)d_in[16];
  const float* w2 = (const float*)d_in[17];
  const float* b2 = (const float*)d_in[18];
  const float* g2 = (const float*)d_in[19];
  const float* be2 = (const float*)d_in[20];
  const float* m2 = (const float*)d_in[21];
  const float* v2 = (const float*)d_in[22];
  float* out = (float*)d_out;

  dim3 grid(BATCH * NSAMP / 4), block(128);

  if (ws_size >= (size_t)WIMG_BYTES) {
    unsigned short* wimg = (unsigned short*)d_ws;
    fold_weights_kernel<<<dim3(1), dim3(256), 0, stream>>>(
        w0, b0, g0, be0, m0, v0,
        w1, b1, g1, be1, m1, v1,
        w2, b2, g2, be2, m2, v2, wimg);
    pointnet_sa_kernel<false><<<grid, block, 0, stream>>>(
        xyz_proj, points_proj, xyz_sampled, neighbor_idx, valid_mask,
        w0, b0, g0, be0, m0, v0,
        w1, b1, g1, be1, m1, v1,
        w2, b2, g2, be2, m2, v2,
        wimg, out);
  } else {
    pointnet_sa_kernel<true><<<grid, block, 0, stream>>>(
        xyz_proj, points_proj, xyz_sampled, neighbor_idx, valid_mask,
        w0, b0, g0, be0, m0, v0,
        w1, b1, g1, be1, m1, v1,
        w2, b2, g2, be2, m2, v2,
        nullptr, out);
  }
}
